// ModifiedBertAttention_65584150610105
// MI455X (gfx1250) — matrix operand order repaired and checked by reading, not compiled, not run on hardware
//
#include <hip/hip_runtime.h>

// B=2, S=2048, C=1024, H=16, D=64
// Matrix ops: v_wmma_f32_16x16x32_bf16 (fp32 accumulate).
// Data movement: global_load_async_to_lds_b128 (ASYNCcnt) when available,
// double-buffered LDS tiles everywhere.

typedef __attribute__((ext_vector_type(16))) __bf16        bf16x16;
typedef __attribute__((ext_vector_type(8)))  float         f32x8;
typedef __attribute__((ext_vector_type(8)))  unsigned int  u32x8;

#if defined(__gfx1250__) && __has_builtin(__builtin_amdgcn_global_load_async_to_lds_b128)
#define USE_ASYNC_LDS 1
typedef int i32x4 __attribute__((ext_vector_type(4)));
typedef __attribute__((address_space(1))) i32x4* as1_i32x4p;   // global
typedef __attribute__((address_space(3))) i32x4* as3_i32x4p;   // LDS
#else
#define USE_ASYNC_LDS 0
#endif

// 16-byte global -> LDS copy (async DMA path on CDNA5, VGPR fallback otherwise)
static __device__ __forceinline__ void cp16(unsigned short* lds, const unsigned short* g) {
#if USE_ASYNC_LDS
    __builtin_amdgcn_global_load_async_to_lds_b128((as1_i32x4p)g, (as3_i32x4p)lds, 0, 0);
#else
    *reinterpret_cast<uint4*>(lds) = *reinterpret_cast<const uint4*>(g);
#endif
}
static __device__ __forceinline__ void cp_wait() {
#if USE_ASYNC_LDS
#if __has_builtin(__builtin_amdgcn_s_wait_asynccnt)
    __builtin_amdgcn_s_wait_asynccnt(0);
#else
    asm volatile("s_wait_asynccnt 0x0" ::: "memory");
#endif
#endif
}

static __device__ __forceinline__ unsigned short f2bf(float f) {
    unsigned int u = __builtin_bit_cast(unsigned int, f);
    u = (u + 0x7FFFu + ((u >> 16) & 1u)) >> 16;   // RNE
    return (unsigned short)u;
}

static __device__ __forceinline__ f32x8 wmma_bf16(bf16x16 a, bf16x16 b, f32x8 c) {
    return __builtin_amdgcn_wmma_f32_16x16x32_bf16(false, a, false, b, (short)0, c, false, false);
}

// A fragment 16x32 bf16 (ISA 7.12.2): lanes 0-15 row M=lane, K 0..7 & 16..23;
// lanes 16-31 same rows, K 8..15 & 24..31.  Two ds_load_b128.
static __device__ __forceinline__ bf16x16 load_frag_a(const unsigned short* base,
                                                      int row0, int k0, int ld, int lane) {
    int m  = row0 + (lane & 15);
    int kh = (lane >> 4) * 8;
    uint4 a = *reinterpret_cast<const uint4*>(base + m * ld + k0 + kh);
    uint4 b = *reinterpret_cast<const uint4*>(base + m * ld + k0 + 16 + kh);
    u32x8 r; r[0]=a.x; r[1]=a.y; r[2]=a.z; r[3]=a.w; r[4]=b.x; r[5]=b.y; r[6]=b.z; r[7]=b.w;
    return __builtin_bit_cast(bf16x16, r);
}

// B fragment 32x16 bf16: lane holds column N = lane&15; K 0..7 & 16..23 (lanes 0-15), K 8..15 & 24..31 (lanes 16-31).
static __device__ __forceinline__ bf16x16 load_frag_b(const unsigned short* base,
                                                      int k0, int n0, int ld, int lane) {
    int n  = n0 + (lane & 15);
    int kh = k0 + (lane >> 4) * 8;
    u32x8 r;
    #pragma unroll
    for (int i = 0; i < 4; ++i) {
        r[i]     = (unsigned int)base[(kh + 2 * i) * ld + n] | ((unsigned int)base[(kh + 2 * i + 1) * ld + n] << 16);
        r[4 + i] = (unsigned int)base[(kh + 16 + 2 * i) * ld + n] | ((unsigned int)base[(kh + 17 + 2 * i) * ld + n] << 16);
    }
    return __builtin_bit_cast(bf16x16, r);
}

// ---------------- Kernel 1: fp32 -> bf16 packing ----------------
__global__ __launch_bounds__(256) void convert_all(
    const float* __restrict__ x,  const float* __restrict__ wq,
    const float* __restrict__ wk, const float* __restrict__ wv,
    const float* __restrict__ wo,
    unsigned short* __restrict__ xb, unsigned short* __restrict__ wqkv,
    unsigned short* __restrict__ wob) {
    int stride = gridDim.x * blockDim.x;
    int i0 = blockIdx.x * blockDim.x + threadIdx.x;
    for (int i = i0; i < 4096 * 1024; i += stride) xb[i] = f2bf(x[i]);
    for (int i = i0; i < 1024 * 1024; i += stride) {
        wqkv[i]           = f2bf(wq[i]);
        wqkv[i + 1048576] = f2bf(wk[i]);
        wqkv[i + 2097152] = f2bf(wv[i]);
        wob[i]            = f2bf(wo[i]);
    }
}

// ---------------- Kernel 2: QKV GEMM (64x128 tile, 4 waves, double-buffered) ----------------
__global__ __launch_bounds__(128) void qkv_gemm(
    const unsigned short* __restrict__ xb, const unsigned short* __restrict__ wqkv,
    const float* __restrict__ bq, const float* __restrict__ bk, const float* __restrict__ bv,
    unsigned short* __restrict__ qo, unsigned short* __restrict__ ko,
    unsigned short* __restrict__ vo) {
    const int g = blockIdx.z;
    const unsigned short* W = wqkv + (size_t)g * 1048576;
    const float* bias = (g == 0) ? bq : (g == 1) ? bk : bv;
    unsigned short* out = (g == 0) ? qo : (g == 1) ? ko : vo;
    const int mblk = blockIdx.y * 64, nblk = blockIdx.x * 128;

    __shared__ alignas(16) unsigned short As[2][64 * 40];
    __shared__ alignas(16) unsigned short Bs[2][32 * 136];

    const int tid = threadIdx.x, lane = tid & 31, w = tid >> 5;
    const int wm = (w >> 1) * 32, wn = (w & 1) * 64;

    const f32x8 zero = {0.f,0.f,0.f,0.f,0.f,0.f,0.f,0.f};
    f32x8 acc[2][4];
    #pragma unroll
    for (int mt = 0; mt < 2; ++mt)
        #pragma unroll
        for (int j = 0; j < 4; ++j) acc[mt][j] = zero;

    auto stage = [&](int kk, int buf) {
        #pragma unroll
        for (int r = 0; r < 2; ++r) {                 // A: 64x32
            int li = tid + r * 128, row = li >> 2, col = (li & 3) * 8;
            cp16(&As[buf][row * 40 + col], xb + (size_t)(mblk + row) * 1024 + kk + col);
        }
        #pragma unroll
        for (int r = 0; r < 4; ++r) {                 // B: 32x128
            int li = tid + r * 128, row = li >> 4, col = (li & 15) * 8;
            cp16(&Bs[buf][row * 136 + col], W + (size_t)(kk + row) * 1024 + nblk + col);
        }
    };

    stage(0, 0);
    cp_wait();
    __syncthreads();

    for (int it = 0; it < 32; ++it) {
        const int cur = it & 1;
        if (it + 1 < 32) stage((it + 1) * 32, cur ^ 1);   // prefetch next tile
        bf16x16 a0 = load_frag_a(As[cur], wm,      0, 40, lane);
        bf16x16 a1 = load_frag_a(As[cur], wm + 16, 0, 40, lane);
        #pragma unroll
        for (int j = 0; j < 4; ++j) {
            bf16x16 bj = load_frag_b(Bs[cur], 0, wn + j * 16, 136, lane);
            acc[0][j] = wmma_bf16(a0, bj, acc[0][j]);
            acc[1][j] = wmma_bf16(a1, bj, acc[1][j]);
        }
        cp_wait();
        __syncthreads();
    }

    #pragma unroll
    for (int j = 0; j < 4; ++j) {
        const float bj = bias[nblk + wn + j * 16 + (lane & 15)];
        #pragma unroll
        for (int mt = 0; mt < 2; ++mt)
            #pragma unroll
            for (int r = 0; r < 8; ++r) {
                int m = mblk + wm + mt * 16 + r + ((lane >> 4) << 3);
                int n = nblk + wn + j * 16 + (lane & 15);
                int b = m >> 11, s = m & 2047, h = n >> 6, d = n & 63;
                out[(((size_t)b * 16 + h) * 2048 + s) * 64 + d] = f2bf(acc[mt][j][r] + bj);
            }
    }
}

// ---------------- Kernel 3: flash attention (double-buffered K/V chunks) ----------------
__global__ __launch_bounds__(256) void attn(
    const unsigned short* __restrict__ qb, const unsigned short* __restrict__ kb,
    const unsigned short* __restrict__ vb, const float* __restrict__ mask,
    unsigned short* __restrict__ ctxo) {
    const int b = blockIdx.z, h = blockIdx.y, qblk = blockIdx.x * 128;
    const size_t head = ((size_t)b * 16 + h) * 2048 * 64;
    const unsigned short* Q = qb + head;
    const unsigned short* K = kb + head;
    const unsigned short* V = vb + head;

    __shared__ alignas(16) unsigned short Qs[128 * 72];
    __shared__ alignas(16) unsigned short Kt[2][64 * 40];   // transposed: [d][key]
    __shared__ alignas(16) unsigned short Vs[2][32 * 72];
    __shared__ alignas(16) unsigned short Ps[8][16 * 32];

    const int tid = threadIdx.x, lane = tid & 31, w = tid >> 5;

    #pragma unroll
    for (int i = 0; i < 4; ++i) {                           // Q tile 128x64
        int li = tid + i * 256, row = li >> 3, col = (li & 7) * 8;
        uint4 v = *reinterpret_cast<const uint4*>(Q + (size_t)(qblk + row) * 64 + col);
        *reinterpret_cast<uint4*>(Qs + row * 72 + col) = v;
    }
    __syncthreads();
    const bf16x16 qf0 = load_frag_a(Qs, w * 16, 0,  72, lane);
    const bf16x16 qf1 = load_frag_a(Qs, w * 16, 32, 72, lane);

    const f32x8 zero = {0.f,0.f,0.f,0.f,0.f,0.f,0.f,0.f};
    f32x8 ctx[4];
    ctx[0] = zero; ctx[1] = zero; ctx[2] = zero; ctx[3] = zero;
    float rmax[8], rsum[8];
    #pragma unroll
    for (int r = 0; r < 8; ++r) { rmax[r] = -1e30f; rsum[r] = 0.f; }

    const float scale = 0.125f;                             // 1/sqrt(64)
    const int qrow_lo = qblk + w * 16 + ((lane >> 4) << 3);
    const float* mbase = mask + ((size_t)b * 2048 + qrow_lo) * 2048;

    const int key = tid >> 3, dcol = (tid & 7) * 8;         // cooperative copy coords

    // ---- prologue: stage chunk 0 ----
    {
        uint4 k0 = *reinterpret_cast<const uint4*>(K + (size_t)key * 64 + dcol);
        cp16(&Vs[0][key * 72 + dcol], V + (size_t)key * 64 + dcol);
        const unsigned short* pv = reinterpret_cast<const unsigned short*>(&k0);
        #pragma unroll
        for (int j = 0; j < 8; ++j) Kt[0][(dcol + j) * 40 + key] = pv[j];
        cp_wait();
        __syncthreads();
    }

    for (int i = 0; i < 64; ++i) {
        const int cur = i & 1;
        const int kb0 = i * 32;
        uint4 knext = {0, 0, 0, 0};
        const bool more = (i + 1) < 64;
        if (more) {                                          // prefetch chunk i+1
            knext = *reinterpret_cast<const uint4*>(K + (size_t)(kb0 + 32 + key) * 64 + dcol);
            cp16(&Vs[cur ^ 1][key * 72 + dcol], V + (size_t)(kb0 + 32 + key) * 64 + dcol);
        }

        f32x8 s0 = zero, s1 = zero;                          // scores 16x32
        s0 = wmma_bf16(qf0, load_frag_b(Kt[cur], 0,  0,  40, lane), s0);
        s0 = wmma_bf16(qf1, load_frag_b(Kt[cur], 32, 0,  40, lane), s0);
        s1 = wmma_bf16(qf0, load_frag_b(Kt[cur], 0,  16, 40, lane), s1);
        s1 = wmma_bf16(qf1, load_frag_b(Kt[cur], 32, 16, 40, lane), s1);

        const int keyb = kb0 + (lane & 15);
        #pragma unroll
        for (int r = 0; r < 8; ++r) {
            s0[r] = s0[r] * scale + mbase[(size_t)r * 2048 + keyb];
            s1[r] = s1[r] * scale + mbase[(size_t)r * 2048 + keyb + 16];
        }
        #pragma unroll
        for (int r = 0; r < 8; ++r) {                        // online softmax per row
            float cm = fmaxf(s0[r], s1[r]);
            #pragma unroll
            for (int m = 8; m >= 1; m >>= 1) cm = fmaxf(cm, __shfl_xor(cm, m, 32));
            float nm = fmaxf(rmax[r], cm);
            float corr = __expf(rmax[r] - nm);
            float p0 = __expf(s0[r] - nm);
            float p1 = __expf(s1[r] - nm);
            float ps = p0 + p1;
            #pragma unroll
            for (int m = 8; m >= 1; m >>= 1) ps += __shfl_xor(ps, m, 32);
            rsum[r] = rsum[r] * corr + ps;
            rmax[r] = nm;
            ctx[0][r] *= corr; ctx[1][r] *= corr; ctx[2][r] *= corr; ctx[3][r] *= corr;
            s0[r] = p0; s1[r] = p1;
        }
        {   // stage P (per-wave LDS region, same-wave ordering via DScnt)
            unsigned short* P = &Ps[w][0];
            int pr = ((lane >> 4) << 3), pc = lane & 15;
            #pragma unroll
            for (int r = 0; r < 8; ++r) {
                P[(pr + r) * 32 + pc]      = f2bf(s0[r]);
                P[(pr + r) * 32 + pc + 16] = f2bf(s1[r]);
            }
        }
        bf16x16 pf = load_frag_a(&Ps[w][0], 0, 0, 32, lane);
        #pragma unroll
        for (int t = 0; t < 4; ++t)
            ctx[t] = wmma_bf16(pf, load_frag_b(Vs[cur], 0, t * 16, 72, lane), ctx[t]);

        if (more) {                                          // transpose K chunk i+1
            const unsigned short* pv = reinterpret_cast<const unsigned short*>(&knext);
            #pragma unroll
            for (int j = 0; j < 8; ++j) Kt[cur ^ 1][(dcol + j) * 40 + key] = pv[j];
        }
        cp_wait();
        __syncthreads();
    }

    #pragma unroll
    for (int t = 0; t < 4; ++t)
        #pragma unroll
        for (int r = 0; r < 8; ++r) {
            float vv = ctx[t][r] / rsum[r];
            int q = qblk + w * 16 + r + ((lane >> 4) << 3);
            int col = h * 64 + t * 16 + (lane & 15);
            ctxo[((size_t)b * 2048 + q) * 1024 + col] = f2bf(vv);   // [B,S,H*D]
        }
}

// ---------------- Kernel 4: output projection + bias + residual ----------------
__global__ __launch_bounds__(128) void out_proj(
    const unsigned short* __restrict__ ctxb, const unsigned short* __restrict__ wob,
    const float* __restrict__ bo, const float* __restrict__ x, float* __restrict__ y) {
    const int mblk = blockIdx.y * 64, nblk = blockIdx.x * 128;
    __shared__ alignas(16) unsigned short As[2][64 * 40];
    __shared__ alignas(16) unsigned short Bs[2][32 * 136];
    const int tid = threadIdx.x, lane = tid & 31, w = tid >> 5;
    const int wm = (w >> 1) * 32, wn = (w & 1) * 64;

    const f32x8 zero = {0.f,0.f,0.f,0.f,0.f,0.f,0.f,0.f};
    f32x8 acc[2][4];
    #pragma unroll
    for (int mt = 0; mt < 2; ++mt)
        #pragma unroll
        for (int j = 0; j < 4; ++j) acc[mt][j] = zero;

    auto stage = [&](int kk, int buf) {
        #pragma unroll
        for (int r = 0; r < 2; ++r) {
            int li = tid + r * 128, row = li >> 2, col = (li & 3) * 8;
            cp16(&As[buf][row * 40 + col], ctxb + (size_t)(mblk + row) * 1024 + kk + col);
        }
        #pragma unroll
        for (int r = 0; r < 4; ++r) {
            int li = tid + r * 128, row = li >> 4, col = (li & 15) * 8;
            cp16(&Bs[buf][row * 136 + col], wob + (size_t)(kk + row) * 1024 + nblk + col);
        }
    };

    stage(0, 0);
    cp_wait();
    __syncthreads();

    for (int it = 0; it < 32; ++it) {
        const int cur = it & 1;
        if (it + 1 < 32) stage((it + 1) * 32, cur ^ 1);
        bf16x16 a0 = load_frag_a(As[cur], wm,      0, 40, lane);
        bf16x16 a1 = load_frag_a(As[cur], wm + 16, 0, 40, lane);
        #pragma unroll
        for (int j = 0; j < 4; ++j) {
            bf16x16 bj = load_frag_b(Bs[cur], 0, wn + j * 16, 136, lane);
            acc[0][j] = wmma_bf16(a0, bj, acc[0][j]);
            acc[1][j] = wmma_bf16(a1, bj, acc[1][j]);
        }
        cp_wait();
        __syncthreads();
    }

    #pragma unroll
    for (int j = 0; j < 4; ++j) {
        const float bj = bo[nblk + wn + j * 16 + (lane & 15)];
        #pragma unroll
        for (int mt = 0; mt < 2; ++mt)
            #pragma unroll
            for (int r = 0; r < 8; ++r) {
                int m = mblk + wm + mt * 16 + r + ((lane >> 4) << 3);
                size_t i0 = (size_t)m * 1024 + nblk + wn + j * 16 + (lane & 15);
                y[i0] = acc[mt][j][r] + bj + x[i0];
            }
    }
}

// ---------------- Kernel 5: LayerNorm over C=1024 ----------------
__global__ __launch_bounds__(256) void layer_norm(
    const float* __restrict__ y, const float* __restrict__ gamma,
    const float* __restrict__ beta, float* __restrict__ out) {
    __shared__ float s1[256];
    __shared__ float s2[256];
    const int row = blockIdx.x, t = threadIdx.x;
    const float* yr = y + (size_t)row * 1024;
    float v[4], a = 0.f, b = 0.f;
    #pragma unroll
    for (int i = 0; i < 4; ++i) { v[i] = yr[t + 256 * i]; a += v[i]; b += v[i] * v[i]; }
    s1[t] = a; s2[t] = b; __syncthreads();
    for (int o = 128; o > 0; o >>= 1) {
        if (t < o) { s1[t] += s1[t + o]; s2[t] += s2[t + o]; }
        __syncthreads();
    }
    const float mu  = s1[0] * (1.f / 1024.f);
    const float var = s2[0] * (1.f / 1024.f) - mu * mu;
    const float inv = rsqrtf(var + 1e-12f);
    #pragma unroll
    for (int i = 0; i < 4; ++i) {
        int c = t + 256 * i;
        out[(size_t)row * 1024 + c] = (v[i] - mu) * inv * gamma[c] + beta[c];
    }
}

extern "C" void kernel_launch(void* const* d_in, const int* in_sizes, int n_in,
                              void* d_out, int out_size, void* d_ws, size_t ws_size,
                              hipStream_t stream) {
    const float* x     = (const float*)d_in[0];
    const float* mask  = (const float*)d_in[1];
    const float* Wq    = (const float*)d_in[2];
    const float* bq    = (const float*)d_in[3];
    const float* Wk    = (const float*)d_in[4];
    const float* bk    = (const float*)d_in[5];
    const float* Wv    = (const float*)d_in[6];
    const float* bv    = (const float*)d_in[7];
    const float* Wo    = (const float*)d_in[8];
    const float* bo    = (const float*)d_in[9];
    const float* gamma = (const float*)d_in[10];
    const float* beta  = (const float*)d_in[11];
    float* out = (float*)d_out;

    // workspace layout (64 MB total, all 16B aligned)
    char* ws = (char*)d_ws;
    unsigned short* xb   = (unsigned short*)(ws);                 // 8 MB  x bf16 [4096,1024]
    unsigned short* wqkv = (unsigned short*)(ws + 8388608);       // 6 MB  Wq|Wk|Wv bf16
    unsigned short* wob  = (unsigned short*)(ws + 14680064);      // 2 MB  Wo bf16
    unsigned short* qb   = (unsigned short*)(ws + 16777216);      // 8 MB  q bf16 [B,H,S,D]
    unsigned short* kbp  = (unsigned short*)(ws + 25165824);      // 8 MB  k
    unsigned short* vbp  = (unsigned short*)(ws + 33554432);      // 8 MB  v
    unsigned short* ctxb = (unsigned short*)(ws + 41943040);      // 8 MB  ctx bf16 [4096,1024]
    float*          yf   = (float*)(ws + 50331648);               // 16 MB y fp32

    convert_all<<<2048, 256, 0, stream>>>(x, Wq, Wk, Wv, Wo, xb, wqkv, wob);
    qkv_gemm<<<dim3(8, 64, 3), 128, 0, stream>>>(xb, wqkv, bq, bk, bv, qb, kbp, vbp);
    attn<<<dim3(16, 16, 2), 256, 0, stream>>>(qb, kbp, vbp, mask, ctxb);
    out_proj<<<dim3(8, 64), 128, 0, stream>>>(ctxb, wob, bo, x, yf);
    layer_norm<<<4096, 256, 0, stream>>>(yf, gamma, beta, out);
}
